// Bah_Attn_54030688584149
// MI455X (gfx1250) — matrix operand order repaired and checked by reading, not compiled, not run on hardware
//
#include <hip/hip_runtime.h>
#include <hip/hip_bf16.h>

// ---------------------------------------------------------------------------
// Bahdanau attention for MI455X (gfx1250, wave32, WMMA).
//   context, score = attn(x, memory, W1,b1, W2,b2, v,bv)
// Big GEMM (memory @ W2, 275 GFLOP) runs as v_wmma_f32_16x16x32_bf16 with the
// tanh/v-dot epilogue fused so h_m is never materialized. Everything else is
// bandwidth-bound f32.
// ---------------------------------------------------------------------------

typedef __attribute__((ext_vector_type(16))) __bf16 v16bf;
typedef __attribute__((ext_vector_type(8)))  __bf16 v8bf;
typedef __attribute__((ext_vector_type(8)))  float  v8f;

#define BSZ   32
#define SLEN  2048
#define INPD  1024
#define HID   1024
#define MEMD  2048
#define KCH   1024   // K-chunk staged in LDS (bf16): 16 rows * 1024 * 2B = 32 KB

// ---------------------------------------------------------------------- h_x
__global__ __launch_bounds__(256)
void hx_kernel(const float* __restrict__ x, const float* __restrict__ W1,
               const float* __restrict__ b1, float* __restrict__ hx) {
  const int b = blockIdx.y;
  const int h = blockIdx.x * 256 + threadIdx.x;
  float acc = b1[h];
  const float* xr = x + b * INPD;
#pragma unroll 4
  for (int k = 0; k < INPD; ++k)
    acc += xr[k] * W1[(size_t)k * HID + h];
  hx[b * HID + h] = acc;
}

// ------------------------------------------------- W2 -> W2T (bf16, [n][k])
__global__ __launch_bounds__(256)
void w2t_kernel(const float* __restrict__ W2, __bf16* __restrict__ W2T) {
  __shared__ float tile[32][33];
  const int k0 = blockIdx.x * 32;          // k in [0,2048)
  const int n0 = blockIdx.y * 32;          // n in [0,1024)
  const int tx = threadIdx.x & 31;
  const int ty = threadIdx.x >> 5;         // 32x8 threads
#pragma unroll
  for (int j = 0; j < 4; ++j) {
    const int kk = ty + j * 8;
    tile[kk][tx] = W2[(size_t)(k0 + kk) * HID + n0 + tx];   // coalesced over n
  }
  __syncthreads();
#pragma unroll
  for (int j = 0; j < 4; ++j) {
    const int nn = ty + j * 8;
    W2T[(size_t)(n0 + nn) * MEMD + k0 + tx] = (__bf16)tile[tx][nn]; // coalesced over k
  }
}

// ---------------------------------------------------------- fused score GEMM
// block = 256 threads (8 waves); one block per (batch, 16-row seq tile).
// wave w owns N-tiles nt = w*8 .. w*8+7 (8 * 16 = 128 hidden cols per wave).
__global__ __launch_bounds__(256)
void score_kernel(const float* __restrict__ mem, const __bf16* __restrict__ W2T,
                  const float* __restrict__ hx,  const float* __restrict__ b2,
                  const float* __restrict__ vw,  const float* __restrict__ bv,
                  float* __restrict__ raw) {
  __shared__ alignas(16) __bf16 smA[16 * KCH];   // 32 KB A chunk (bf16)
  __shared__ float smP[8 * 16];                  // per-wave score partials

  const int b    = blockIdx.y;
  const int s0   = blockIdx.x * 16;
  const int tid  = threadIdx.x;
  const int w    = tid >> 5;
  const int lane = tid & 31;
  const int lm   = lane & 15;
  const int half = lane >> 4;

  const float* memBase = mem + (size_t)(b * SLEN + s0) * MEMD;

  v8f c[8];
  {
    v8f z = {};
#pragma unroll
    for (int i = 0; i < 8; ++i) c[i] = z;
  }

  for (int kc = 0; kc < MEMD / KCH; ++kc) {
    // ---- cooperative stage: 16 rows x KCH f32 -> bf16 in LDS ----
#pragma unroll
    for (int i = 0; i < (16 * KCH / 4) / 256; ++i) {    // 16 iters of float4
      const int f4  = i * 256 + tid;
      const int row = f4 >> 8;                          // 256 float4 per row
      const int c4  = f4 & 255;
      const float* gsrc = memBase + (size_t)row * MEMD + kc * KCH + c4 * 4;
      const float4 val = *(const float4*)gsrc;
      if (kc == 0) __builtin_prefetch(gsrc + KCH, 0, 1); // next chunk -> global_prefetch_b8
      union { __bf16 h[4]; uint2 u; } p;
      p.h[0] = (__bf16)val.x; p.h[1] = (__bf16)val.y;
      p.h[2] = (__bf16)val.z; p.h[3] = (__bf16)val.w;
      *(uint2*)(&smA[row * KCH + c4 * 4]) = p.u;
    }
    __syncthreads();

    // ---- WMMA: for each owned N-tile, sweep this K chunk ----
    // A layout (16-bit 16x32): lane lm = row m; elems 0..15 -> K
    // k0+16*half+0..15 (contiguous, matches B)  => two 16B LDS loads.
    // B layout (16-bit 32x16): lane -> col n=lm, 16 contiguous K at
    // k0 + 16*half  => one 32B global load from W2T[n][*].
#pragma unroll
    for (int nti = 0; nti < 8; ++nti) {
      const int nt = w * 8 + nti;
      const int n  = nt * 16 + lm;
      const __bf16* bptr = W2T + (size_t)n * MEMD + kc * KCH + half * 16;
      const __bf16* arow = smA + lm * KCH + half * 16;
      v8f acc = c[nti];
#pragma unroll 8
      for (int ks = 0; ks < KCH / 32; ++ks) {
        const int k0 = ks * 32;
        union { v16bf v; v8bf h[2]; } A;
        A.h[0] = *(const v8bf*)(arow + k0);
        A.h[1] = *(const v8bf*)(arow + k0 + 8);
        const v16bf B = *(const v16bf*)(bptr + k0);
        acc = __builtin_amdgcn_wmma_f32_16x16x32_bf16(
            /*neg_a=*/false, A.v, /*neg_b=*/false, B,
            /*c_mod=*/(short)0, acc, /*reuse_a=*/false, /*reuse_b=*/false);
      }
      c[nti] = acc;
    }
    __syncthreads();
  }

  // ---- fused epilogue: part[m] += v[n] * tanh(hx[b,n] + b2[n] + C[m,n]) ----
  // C layout: VGPR r, lanes 0-15 -> M=r, lanes 16-31 -> M=8+r; N = lm.
  float part[8];
#pragma unroll
  for (int r = 0; r < 8; ++r) part[r] = 0.0f;
#pragma unroll
  for (int nti = 0; nti < 8; ++nti) {
    const int nt = w * 8 + nti;
    const int n  = nt * 16 + lm;
    const float hb = hx[b * HID + n] + b2[n];
    const float vn = vw[n];
#pragma unroll
    for (int r = 0; r < 8; ++r)
      part[r] += vn * tanhf(hb + c[nti][r]);
  }
  // reduce over the 16 lanes of each half (xor masks 1..8 stay within halves)
#pragma unroll
  for (int msk = 8; msk >= 1; msk >>= 1)
#pragma unroll
    for (int r = 0; r < 8; ++r)
      part[r] += __shfl_xor(part[r], msk, 32);
  if (lm == 0) {
#pragma unroll
    for (int r = 0; r < 8; ++r)
      smP[w * 16 + half * 8 + r] = part[r];   // m = half*8 + r
  }
  __syncthreads();
  if (tid < 16) {
    float s = 0.0f;
#pragma unroll
    for (int ww = 0; ww < 8; ++ww) s += smP[ww * 16 + tid];
    raw[b * SLEN + s0 + tid] = s + bv[0];
  }
}

// ------------------------------------------------------------------ softmax
__global__ __launch_bounds__(256)
void softmax_kernel(const float* __restrict__ raw, float* __restrict__ prob) {
  __shared__ float red[256];
  const int b = blockIdx.x, tid = threadIdx.x;
  const float* r = raw + b * SLEN;
  float vals[SLEN / 256];
  float mx = -__builtin_inff();
#pragma unroll
  for (int i = 0; i < SLEN / 256; ++i) {
    vals[i] = r[tid + i * 256];
    mx = fmaxf(mx, vals[i]);
  }
  red[tid] = mx; __syncthreads();
  for (int s = 128; s > 0; s >>= 1) {
    if (tid < s) red[tid] = fmaxf(red[tid], red[tid + s]);
    __syncthreads();
  }
  mx = red[0]; __syncthreads();
  float sum = 0.0f;
#pragma unroll
  for (int i = 0; i < SLEN / 256; ++i) {
    vals[i] = __expf(vals[i] - mx);
    sum += vals[i];
  }
  red[tid] = sum; __syncthreads();
  for (int s = 128; s > 0; s >>= 1) {
    if (tid < s) red[tid] += red[tid + s];
    __syncthreads();
  }
  const float inv = 1.0f / red[0];
#pragma unroll
  for (int i = 0; i < SLEN / 256; ++i)
    prob[b * SLEN + tid + i * 256] = vals[i] * inv;
}

// --------------------------------------------------- context = prob @ memory
__global__ __launch_bounds__(256)
void context_kernel(const float* __restrict__ mem, const float* __restrict__ prob,
                    float* __restrict__ ctx) {
  __shared__ float p[SLEN];
  const int b = blockIdx.y;
  const int d = blockIdx.x * 256 + threadIdx.x;
#pragma unroll
  for (int i = 0; i < SLEN / 256; ++i)
    p[threadIdx.x + i * 256] = prob[b * SLEN + threadIdx.x + i * 256];
  __syncthreads();
  const float* m = mem + (size_t)b * SLEN * MEMD + d;
  float acc = 0.0f;
#pragma unroll 4
  for (int s = 0; s < SLEN; ++s)
    acc += p[s] * m[(size_t)s * MEMD];
  ctx[b * MEMD + d] = acc;
}

// ---------------------------------------------------------------------------
extern "C" void kernel_launch(void* const* d_in, const int* in_sizes, int n_in,
                              void* d_out, int out_size, void* d_ws, size_t ws_size,
                              hipStream_t stream) {
  (void)in_sizes; (void)n_in; (void)out_size; (void)ws_size;
  const float* x      = (const float*)d_in[0];
  const float* memory = (const float*)d_in[1];
  const float* W1     = (const float*)d_in[2];
  const float* b1     = (const float*)d_in[3];
  const float* W2     = (const float*)d_in[4];
  const float* b2     = (const float*)d_in[5];
  const float* v      = (const float*)d_in[6];
  const float* bv     = (const float*)d_in[7];

  // workspace layout
  char* ws = (char*)d_ws;
  __bf16* W2T   = (__bf16*)ws;                                  // 4 MB
  float*  hx    = (float*)(ws + (size_t)MEMD * HID * 2);        // 128 KB
  float*  raw   = (float*)(ws + (size_t)MEMD * HID * 2 + (size_t)BSZ * HID * 4); // 256 KB

  float* out_ctx   = (float*)d_out;                 // [32, 2048]
  float* out_score = (float*)d_out + BSZ * MEMD;    // [32, 2048]

  hx_kernel   <<<dim3(HID / 256, BSZ),  256, 0, stream>>>(x, W1, b1, hx);
  w2t_kernel  <<<dim3(MEMD / 32, HID / 32), 256, 0, stream>>>(W2, W2T);
  score_kernel<<<dim3(SLEN / 16, BSZ),  256, 0, stream>>>(memory, W2T, hx, b2, v, bv, raw);
  softmax_kernel<<<BSZ, 256, 0, stream>>>(raw, out_score);
  context_kernel<<<dim3(MEMD / 256, BSZ), 256, 0, stream>>>(memory, out_score, out_ctx);
}
